// GLBlock_10058813407564
// MI455X (gfx1250) — compile-verified, matrix operand order checked by reading, not run on hardware
//
#include <hip/hip_runtime.h>
#include <hip/hip_bf16.h>
#include <math.h>
#include <stdint.h>

typedef unsigned short u16;
typedef unsigned int   u32;
typedef __attribute__((ext_vector_type(16))) __bf16 v16bf;
typedef __attribute__((ext_vector_type(8)))  float  v8f;

#define Bc   4
#define Tc   2048
#define Dc   768
#define Hc   12
#define DHc  64
#define FFc  3072
#define Mrows (Bc*Tc)      // 8192
#define RAD  64

#if defined(__gfx1250__) && __has_builtin(__builtin_amdgcn_global_load_async_to_lds_b128) && __has_builtin(__builtin_amdgcn_s_wait_asynccnt)
#define USE_ASYNC 1
#else
#define USE_ASYNC 0
#endif

__device__ __forceinline__ u16 bf16_rne(float f) {
  u32 u = __float_as_uint(f);
  u += 0x7fffu + ((u >> 16) & 1u);
  return (u16)(u >> 16);
}

#if USE_ASYNC
// Direct global->LDS async copy of 16 bytes per lane.  Flat pointers to
// __shared__ carry the LDS byte offset in their low 32 bits (aperture rule),
// so the integer round-trip yields a valid AS(3) pointer; AS(1) keeps the
// full 64-bit address.  Builtin expects non-const pointers to
// 'int __attribute__((vector_size(16)))' in AS1 (global) / AS3 (LDS).
typedef int v4i_vs __attribute__((vector_size(16)));
__device__ __forceinline__ void cp16_g2l(void* lds, const void* g) {
  __builtin_amdgcn_global_load_async_to_lds_b128(
      (__attribute__((address_space(1))) v4i_vs*)(unsigned long long)(uintptr_t)g,
      (__attribute__((address_space(3))) v4i_vs*)(unsigned int)(uintptr_t)lds,
      0, 0);
}
#endif

union FragBF {
  uint4 q[2];
  v16bf v;
};

// ---------------------------------------------------------------------------
// Weight transpose + fp32->bf16 convert:  W[Kd][Nd] -> dst[Nd][Kd] (bf16)
// ---------------------------------------------------------------------------
__global__ __launch_bounds__(256) void wconv(const float* __restrict__ W,
                                             u16* __restrict__ dst,
                                             int Kd, int Nd) {
  int e = blockIdx.x * 256 + threadIdx.x;
  if (e >= Kd * Nd) return;
  int n = e / Kd;
  int k = e - n * Kd;
  dst[e] = bf16_rne(W[(size_t)k * Nd + n]);
}

// ---------------------------------------------------------------------------
// LayerNorm over rows of 768, output bf16. One block (256 thr) per row.
// ---------------------------------------------------------------------------
__global__ __launch_bounds__(256) void ln_kernel(const float* __restrict__ x,
                                                 const float* __restrict__ g,
                                                 const float* __restrict__ be,
                                                 u16* __restrict__ out) {
  __shared__ float sA[256];
  __shared__ float sB[256];
  int row = blockIdx.x;
  int tid = threadIdx.x;
  const float* xr = x + (size_t)row * Dc;
  float v0 = xr[tid], v1 = xr[tid + 256], v2 = xr[tid + 512];
  sA[tid] = v0 + v1 + v2;
  sB[tid] = v0 * v0 + v1 * v1 + v2 * v2;
  __syncthreads();
  for (int o = 128; o > 0; o >>= 1) {
    if (tid < o) { sA[tid] += sA[tid + o]; sB[tid] += sB[tid + o]; }
    __syncthreads();
  }
  float mu  = sA[0] * (1.0f / Dc);
  float var = sB[0] * (1.0f / Dc) - mu * mu;
  float rinv = rsqrtf(var + 1e-5f);
  u16* orow = out + (size_t)row * Dc;
  orow[tid]       = bf16_rne((v0 - mu) * rinv * g[tid]       + be[tid]);
  orow[tid + 256] = bf16_rne((v1 - mu) * rinv * g[tid + 256] + be[tid + 256]);
  orow[tid + 512] = bf16_rne((v2 - mu) * rinv * g[tid + 512] + be[tid + 512]);
}

// ---------------------------------------------------------------------------
// WMMA bf16 GEMM:  C[M,N] = A[M,K] * Bt[N,K]^T  (+ epilogue)
// Block tile 128x128, K-step 32, 256 threads = 8 waves, each wave 32x64.
// LDS double-buffered; tiles streamed with async global->LDS when available.
// MODE 0: QKV  -> scatter fp32 to [B,H,T,DH] with per-segment bias
// MODE 1: fp32 out = acc + bias + residual
// MODE 2: bf16 out = gelu(acc + bias)
// ---------------------------------------------------------------------------
template <int MODE>
__global__ __launch_bounds__(256)
void gemm_bf16(const u16* __restrict__ A, const u16* __restrict__ Bt,
               int M, int N, int K,
               const float* __restrict__ b0, const float* __restrict__ b1,
               const float* __restrict__ b2,
               const float* __restrict__ res,
               float* __restrict__ outF, u16* __restrict__ outB,
               float* __restrict__ oQ, float* __restrict__ oK,
               float* __restrict__ oV) {
  __shared__ __align__(16) u16 As[2][128 * 40];
  __shared__ __align__(16) u16 Bs[2][128 * 40];

  const int tid   = threadIdx.x;
  const int lane  = tid & 31;
  const int wave  = tid >> 5;
  const int waveM = wave & 3;    // 4 row-groups of 32
  const int waveN = wave >> 2;   // 2 col-groups of 64
  const int m16   = lane & 15;
  const int half  = lane >> 4;
  const int blockN = blockIdx.x * 128;
  const int blockM = blockIdx.y * 128;

  v8f acc[2][4];
#pragma unroll
  for (int i = 0; i < 2; i++)
#pragma unroll
    for (int j = 0; j < 4; j++)
#pragma unroll
      for (int e = 0; e < 8; e++) acc[i][j][e] = 0.0f;

  const int rowL = tid >> 2;   // 0..63
  const int chL  = tid & 3;    // 16B chunk within a 64B row
  const int steps = K >> 5;

#if USE_ASYNC
  auto issue = [&](int buf, int kk) {
    cp16_g2l(&As[buf][rowL * 40 + chL * 8],
             A + (size_t)(blockM + rowL) * K + kk + chL * 8);
    cp16_g2l(&As[buf][(64 + rowL) * 40 + chL * 8],
             A + (size_t)(blockM + 64 + rowL) * K + kk + chL * 8);
    cp16_g2l(&Bs[buf][rowL * 40 + chL * 8],
             Bt + (size_t)(blockN + rowL) * K + kk + chL * 8);
    cp16_g2l(&Bs[buf][(64 + rowL) * 40 + chL * 8],
             Bt + (size_t)(blockN + 64 + rowL) * K + kk + chL * 8);
  };
  issue(0, 0);
  __builtin_amdgcn_s_wait_asynccnt(0);
  __syncthreads();
#endif

  for (int i = 0; i < steps; i++) {
#if USE_ASYNC
    const int cur = i & 1;
    if (i + 1 < steps) issue(cur ^ 1, (i + 1) * 32);
#else
    const int cur = 0;
    const int kk = i * 32;
    *reinterpret_cast<uint4*>(&As[0][rowL * 40 + chL * 8]) =
        *reinterpret_cast<const uint4*>(A + (size_t)(blockM + rowL) * K + kk + chL * 8);
    *reinterpret_cast<uint4*>(&As[0][(64 + rowL) * 40 + chL * 8]) =
        *reinterpret_cast<const uint4*>(A + (size_t)(blockM + 64 + rowL) * K + kk + chL * 8);
    *reinterpret_cast<uint4*>(&Bs[0][rowL * 40 + chL * 8]) =
        *reinterpret_cast<const uint4*>(Bt + (size_t)(blockN + rowL) * K + kk + chL * 8);
    *reinterpret_cast<uint4*>(&Bs[0][(64 + rowL) * 40 + chL * 8]) =
        *reinterpret_cast<const uint4*>(Bt + (size_t)(blockN + 64 + rowL) * K + kk + chL * 8);
    __syncthreads();
#endif

    FragBF fa[2], fb[4];
#pragma unroll
    for (int mt = 0; mt < 2; mt++) {
      int r = waveM * 32 + mt * 16 + m16;
      fa[mt].q[0] = *reinterpret_cast<const uint4*>(&As[cur][r * 40 + half * 8]);
      fa[mt].q[1] = *reinterpret_cast<const uint4*>(&As[cur][r * 40 + 16 + half * 8]);
    }
#pragma unroll
    for (int nt = 0; nt < 4; nt++) {
      int r = waveN * 64 + nt * 16 + m16;
      fb[nt].q[0] = *reinterpret_cast<const uint4*>(&Bs[cur][r * 40 + half * 8]);
      fb[nt].q[1] = *reinterpret_cast<const uint4*>(&Bs[cur][r * 40 + 16 + half * 8]);
    }
#pragma unroll
    for (int mt = 0; mt < 2; mt++)
#pragma unroll
      for (int nt = 0; nt < 4; nt++)
        acc[mt][nt] = __builtin_amdgcn_wmma_f32_16x16x32_bf16(
            false, fa[mt].v, false, fb[nt].v, (short)0, acc[mt][nt], false, false);

#if USE_ASYNC
    __builtin_amdgcn_s_wait_asynccnt(0);
#endif
    __syncthreads();
  }

  // Epilogue.  C tile layout: VGPR r, lanes 0-15 -> M=r, lanes 16-31 -> M=r+8;
  // N = lane%16.
#pragma unroll
  for (int mt = 0; mt < 2; mt++) {
#pragma unroll
    for (int nt = 0; nt < 4; nt++) {
      int nc = blockN + waveN * 64 + nt * 16 + m16;
#pragma unroll
      for (int r = 0; r < 8; r++) {
        int mr = blockM + waveM * 32 + mt * 16 + r + half * 8;
        float val = acc[mt][nt][r];
        if (MODE == 0) {
          int seg = nc / Dc;
          int nn  = nc - seg * Dc;
          int head = nn >> 6, dh = nn & 63;
          int bb = mr >> 11;        // / T (2048)
          int tt = mr & 2047;
          const float* bp = (seg == 0) ? b0 : (seg == 1) ? b1 : b2;
          float* dst      = (seg == 0) ? oQ : (seg == 1) ? oK : oV;
          dst[(((size_t)bb * Hc + head) * Tc + tt) * DHc + dh] = val + bp[nn];
        } else if (MODE == 1) {
          val += b0[nc] + res[(size_t)mr * N + nc];
          outF[(size_t)mr * N + nc] = val;
        } else {
          val += b0[nc];
          val = 0.5f * val * (1.0f + erff(val * 0.70710678118f));
          outB[(size_t)mr * N + nc] = bf16_rne(val);
        }
      }
    }
  }
}

// ---------------------------------------------------------------------------
// Local attention, radius 64.  One block per (b, h, 64-query tile).
// K window (192 x 64) staged in LDS (pad 65 -> conflict-free); V streamed
// from global (coalesced).  8 waves, each wave handles 8 queries serially.
// ctx written as bf16 [B*T, D] for the Wo GEMM.
// ---------------------------------------------------------------------------
__global__ __launch_bounds__(256)
void attn_kernel(const float* __restrict__ Q, const float* __restrict__ K,
                 const float* __restrict__ V, u16* __restrict__ ctx) {
  __shared__ float Ks[192 * 65];
  __shared__ float Qw[8][68];

  const int ntiles = Tc / 64;
  int bh = blockIdx.x / ntiles;
  int t0 = (blockIdx.x - bh * ntiles) * 64;
  int kbase = t0 - RAD;  // LDS row r <-> key j = kbase + r, r in [0,192)

  const float* Qp = Q + (size_t)bh * Tc * DHc;
  const float* Kp = K + (size_t)bh * Tc * DHc;
  const float* Vp = V + (size_t)bh * Tc * DHc;

  int tid = threadIdx.x;
  for (int e = tid; e < 192 * 64; e += 256) {
    int r = e >> 6, c = e & 63;
    int j = kbase + r;
    if (j >= 0 && j < Tc) Ks[r * 65 + c] = Kp[(size_t)j * DHc + c];
  }
  __syncthreads();

  int wave = tid >> 5, lane = tid & 31;
  int b  = bh / Hc;
  int hh = bh - b * Hc;

  for (int qi = 0; qi < 8; qi++) {
    int t = t0 + wave * 8 + qi;
    // stage q (pre-scaled by 1/sqrt(DH)) into per-wave LDS row
    Qw[wave][lane]      = Qp[(size_t)t * DHc + lane] * 0.125f;
    Qw[wave][lane + 32] = Qp[(size_t)t * DHc + 32 + lane] * 0.125f;

    int jlo = t - RAD; if (jlo < 0) jlo = 0;
    int jhi = t + RAD; if (jhi > Tc - 1) jhi = Tc - 1;
    int rlo = jlo - kbase, rhi = jhi - kbase;

    float s[5];
#pragma unroll
    for (int c5 = 0; c5 < 5; c5++) {
      int r = rlo + c5 * 32 + lane;
      float a = -1e30f;
      if (r <= rhi) {
        a = 0.0f;
        const float* kr = &Ks[r * 65];
        const float* qv = Qw[wave];
#pragma unroll 8
        for (int d = 0; d < 64; d++) a += qv[d] * kr[d];
      }
      s[c5] = a;
    }
    float m = s[0];
#pragma unroll
    for (int c5 = 1; c5 < 5; c5++) m = fmaxf(m, s[c5]);
    for (int o = 16; o > 0; o >>= 1) m = fmaxf(m, __shfl_xor(m, o, 32));

    float p[5], sum = 0.0f;
#pragma unroll
    for (int c5 = 0; c5 < 5; c5++) {
      p[c5] = (s[c5] > -1e29f) ? __expf(s[c5] - m) : 0.0f;
      sum += p[c5];
    }
    for (int o = 16; o > 0; o >>= 1) sum += __shfl_xor(sum, o, 32);

    float c0 = 0.0f, c1 = 0.0f;
#pragma unroll
    for (int c5 = 0; c5 < 5; c5++) {
      for (int l = 0; l < 32; l++) {
        float pj = __shfl(p[c5], l, 32);
        int r = rlo + c5 * 32 + l;
        if (r <= rhi) {
          int j = kbase + r;
          c0 += pj * Vp[(size_t)j * DHc + lane];
          c1 += pj * Vp[(size_t)j * DHc + 32 + lane];
        }
      }
    }
    float inv = 1.0f / sum;
    size_t orow = ((size_t)b * Tc + t) * Dc + hh * DHc;
    ctx[orow + lane]      = bf16_rne(c0 * inv);
    ctx[orow + 32 + lane] = bf16_rne(c1 * inv);
  }
}

// ---------------------------------------------------------------------------
extern "C" void kernel_launch(void* const* d_in, const int* in_sizes, int n_in,
                              void* d_out, int out_size, void* d_ws, size_t ws_size,
                              hipStream_t stream) {
  const float* x   = (const float*)d_in[0];
  const float* Wq  = (const float*)d_in[1];
  const float* bq  = (const float*)d_in[2];
  const float* Wk  = (const float*)d_in[3];
  const float* bk  = (const float*)d_in[4];
  const float* Wv  = (const float*)d_in[5];
  const float* bv  = (const float*)d_in[6];
  const float* Wo  = (const float*)d_in[7];
  const float* bo  = (const float*)d_in[8];
  const float* W1  = (const float*)d_in[9];
  const float* b1  = (const float*)d_in[10];
  const float* W2  = (const float*)d_in[11];
  const float* b2  = (const float*)d_in[12];
  const float* g1  = (const float*)d_in[13];
  const float* be1 = (const float*)d_in[14];
  const float* g2  = (const float*)d_in[15];
  const float* be2 = (const float*)d_in[16];
  float* out = (float*)d_out;
  (void)in_sizes; (void)n_in; (void)out_size; (void)ws_size;

  char* w = (char*)d_ws;
  size_t off = 0;
  auto alloc = [&](size_t bytes) {
    void* p = w + off;
    off = (off + bytes + 255) & ~(size_t)255;
    return p;
  };

  u16*   WqkvT = (u16*)alloc((size_t)2304 * 768 * 2);
  u16*   WoT   = (u16*)alloc((size_t)768 * 768 * 2);
  u16*   W1T   = (u16*)alloc((size_t)3072 * 768 * 2);
  u16*   W2T   = (u16*)alloc((size_t)768 * 3072 * 2);
  u16*   h1    = (u16*)alloc((size_t)Mrows * Dc * 2);
  float* Qb    = (float*)alloc((size_t)Mrows * Dc * 4);
  float* Kb    = (float*)alloc((size_t)Mrows * Dc * 4);
  float* Vb    = (float*)alloc((size_t)Mrows * Dc * 4);
  u16*   ctx   = (u16*)alloc((size_t)Mrows * Dc * 2);
  float* x2    = (float*)alloc((size_t)Mrows * Dc * 4);
  u16*   h2    = (u16*)alloc((size_t)Mrows * Dc * 2);
  u16*   uff   = (u16*)alloc((size_t)Mrows * FFc * 2);

  // weights -> bf16, transposed to [N][K]; Q/K/V packed along N
  wconv<<<(768 * 768 + 255) / 256, 256, 0, stream>>>(Wq, WqkvT,                 768, 768);
  wconv<<<(768 * 768 + 255) / 256, 256, 0, stream>>>(Wk, WqkvT + 768 * 768,     768, 768);
  wconv<<<(768 * 768 + 255) / 256, 256, 0, stream>>>(Wv, WqkvT + 2 * 768 * 768, 768, 768);
  wconv<<<(768 * 768 + 255) / 256, 256, 0, stream>>>(Wo, WoT, 768, 768);
  wconv<<<(768 * 3072 + 255) / 256, 256, 0, stream>>>(W1, W1T, 768, 3072);
  wconv<<<(3072 * 768 + 255) / 256, 256, 0, stream>>>(W2, W2T, 3072, 768);

  // h1 = LN(x; g1, be1)
  ln_kernel<<<Mrows, 256, 0, stream>>>(x, g1, be1, h1);

  // Q,K,V = h1 @ Wqkv + b  (one fused GEMM, N = 2304)
  gemm_bf16<0><<<dim3(2304 / 128, Mrows / 128), 256, 0, stream>>>(
      h1, WqkvT, Mrows, 2304, 768, bq, bk, bv,
      nullptr, nullptr, nullptr, Qb, Kb, Vb);

  // local attention -> ctx (bf16)
  attn_kernel<<<Bc * Hc * (Tc / 64), 256, 0, stream>>>(Qb, Kb, Vb, ctx);

  // x2 = x + ctx @ Wo + bo
  gemm_bf16<1><<<dim3(768 / 128, Mrows / 128), 256, 0, stream>>>(
      ctx, WoT, Mrows, 768, 768, bo, nullptr, nullptr,
      x, x2, nullptr, nullptr, nullptr, nullptr);

  // h2 = LN(x2; g2, be2)
  ln_kernel<<<Mrows, 256, 0, stream>>>(x2, g2, be2, h2);

  // uff = gelu(h2 @ W1 + b1)  (bf16)
  gemm_bf16<2><<<dim3(3072 / 128, Mrows / 128), 256, 0, stream>>>(
      h2, W1T, Mrows, 3072, 768, b1, nullptr, nullptr,
      nullptr, nullptr, uff, nullptr, nullptr, nullptr);

  // out = x2 + uff @ W2 + b2
  gemm_bf16<1><<<dim3(768 / 128, Mrows / 128), 256, 0, stream>>>(
      uff, W2T, Mrows, 768, 3072, b2, nullptr, nullptr,
      x2, out, nullptr, nullptr, nullptr, nullptr);
}
